// LWMSA_82068235092121
// MI455X (gfx1250) — matrix operand order repaired and checked by reading, not compiled, not run on hardware
//
#include <hip/hip_runtime.h>
#include <hip/hip_bf16.h>

typedef __attribute__((ext_vector_type(16))) __bf16 v16bf;
typedef __attribute__((ext_vector_type(8)))  float  v8f;

#define N_TOK 64
#define DIMC  256
#define HEADS 8
#define HD    32

__device__ inline v8f vzero8() {
    v8f z;
#pragma unroll
    for (int i = 0; i < 8; ++i) z[i] = 0.0f;
    return z;
}

__device__ inline v8f wmma_bf16(v16bf a, v16bf b, v8f c) {
    return __builtin_amdgcn_wmma_f32_16x16x32_bf16(
        /*neg_a=*/false, a, /*neg_b=*/false, b,
        /*c_mod=*/(short)0, c, /*reuse_a=*/false, /*reuse_b=*/false);
}

// A tile (16x32 bf16), source stored row-major [M][K] (K contiguous), ld = row stride.
// Contiguous k order used here: lane m = lane&15; elem i -> K = half*16 + i (16 contiguous K per lane)
__device__ inline v16bf load_A_lds(const __bf16* tile, int ld, int lane) {
    const int m = lane & 15, half = lane >> 4;
    const __bf16* r0 = tile + m * ld + half * 16;
    const __bf16* r1 = tile + m * ld + half * 16 + 8;
    v16bf a;
#pragma unroll
    for (int e = 0; e < 8; ++e) { a[e] = r0[e]; a[8 + e] = r1[e]; }
    return a;
}

// B tile (32x16), logical B[k][n] where source is stored [N][K] (K contiguous): B[k][n] = src[n*ld + k]
// Contiguous k order used here: lane n = lane&15; elem i -> K = half*16 + i (16 contiguous K per lane)
__device__ inline v16bf load_B_nk(const __bf16* tile, int ld, int lane) {
    const int n = lane & 15, half = lane >> 4;
    const __bf16* p = tile + n * ld + half * 16;
    v16bf b;
#pragma unroll
    for (int e = 0; e < 16; ++e) b[e] = p[e];
    return b;
}

// Same but source is f32 in global memory (weights), converted to bf16 on load.
__device__ inline v16bf load_B_nk_f32(const float* tile, int ld, int lane) {
    const int n = lane & 15, half = lane >> 4;
    const float* p = tile + (size_t)n * ld + half * 16;
    v16bf b;
#pragma unroll
    for (int e = 0; e < 16; ++e) b[e] = (__bf16)p[e];
    return b;
}

// B tile (32x16), source stored [K][N] (N contiguous), ld = row stride: B[k][n] = src[k*ld + n]
__device__ inline v16bf load_B_kn(const __bf16* tile, int ld, int lane) {
    const int n = lane & 15, half = lane >> 4;
    const __bf16* p = tile + (half * 16) * ld + n;
    v16bf b;
#pragma unroll
    for (int e = 0; e < 16; ++e) b[e] = p[e * ld];
    return b;
}

// Fused windowed MHA: one block = one window (64 tokens), 8 waves = 8 heads.
__global__ __launch_bounds__(256, 1)
void winattn_fused(const float* __restrict__ x,
                   const float* __restrict__ attn_mask,
                   const float* __restrict__ qkv_w,
                   const float* __restrict__ qkv_b,
                   const float* __restrict__ rel_table,
                   const float* __restrict__ proj_w,
                   const float* __restrict__ proj_b,
                   float* __restrict__ out,
                   int nW) {
    extern __shared__ char smem[];
    __bf16* xs  = (__bf16*)(smem);            // [64][256] bf16; reused as attn-out y
    __bf16* q_s = (__bf16*)(smem + 32768);    // [8][64][32] bf16 (pre-scaled q)
    __bf16* k_s = (__bf16*)(smem + 65536);    // [8][64][32]
    __bf16* v_s = (__bf16*)(smem + 98304);    // [8][64][32]
    float*  mk  = (float*) (smem + 131072);   // [64][64] f32 shifted-window mask
    __bf16* p_s = (__bf16*)(smem + 147456);   // [8][16][64] bf16 per-wave P scratch

    const int w    = blockIdx.x;
    const int tid  = threadIdx.x;
    const int wv   = tid >> 5;
    const int lane = tid & 31;
    const int nloc = lane & 15;
    const int half = lane >> 4;

    const float* xw = x + (size_t)w * (N_TOK * DIMC);
    const float* mw = attn_mask + (size_t)(w % nW) * (N_TOK * N_TOK);

    // ---- stage x (f32 -> bf16) and mask into LDS ----
    for (int i = tid; i < N_TOK * DIMC; i += 256) xs[i] = (__bf16)xw[i];
    for (int i = tid; i < N_TOK * N_TOK; i += 256) mk[i] = mw[i];
    __syncthreads();

    // ---- QKV GEMM: [64 x 256] @ [256 x 768] ; wave wv owns output cols [wv*96, wv*96+96) ----
    for (int ctile = 0; ctile < 6; ++ctile) {
        const int ob = wv * 96 + ctile * 16;
        v8f acc[4];
#pragma unroll
        for (int mt = 0; mt < 4; ++mt) acc[mt] = vzero8();
        for (int ks = 0; ks < 8; ++ks) {
            const int kb = ks * 32;
            // B[k][n] = qkv_w[(ob+n)*256 + kb + k]  (einsum bnc,oc->bno)
            v16bf b = load_B_nk_f32(qkv_w + (size_t)ob * DIMC + kb, DIMC, lane);
#pragma unroll
            for (int mt = 0; mt < 4; ++mt) {
                v16bf a = load_A_lds(xs + (mt * 16) * DIMC + kb, DIMC, lane);
                acc[mt] = wmma_bf16(a, b, acc[mt]);
            }
        }
        // epilogue: +bias, scale q, scatter to per-head q/k/v LDS
        const int o  = ob + nloc;
        const float bias = qkv_b[o];
        const int s = o >> 8;            // 0=q 1=k 2=v
        const int h = (o >> 5) & 7;
        const int d = o & 31;
        const float sc = (s == 0) ? 0.17677669529663687f : 1.0f;  // 32^-0.5 on q
        __bf16* dst = (s == 0) ? q_s : (s == 1) ? k_s : v_s;
#pragma unroll
        for (int mt = 0; mt < 4; ++mt)
#pragma unroll
            for (int r = 0; r < 8; ++r) {
                const int m = mt * 16 + r + 8 * half;
                dst[(h * N_TOK + m) * HD + d] = (__bf16)((acc[mt][r] + bias) * sc);
            }
    }
    __syncthreads();

    // ---- attention: wave wv == head h ----
    {
        const int h = wv;
        const __bf16* qh = q_s + h * N_TOK * HD;
        const __bf16* kh = k_s + h * N_TOK * HD;
        const __bf16* vh = v_s + h * N_TOK * HD;
        __bf16* ph = p_s + h * 16 * 64;

        // preload K^T B-tiles (S = q k^T : B[d][nk] = k[nk][d]) and V B-tiles (B[n][d] = v[n][d])
        v16bf kbt[4], vbt[2][2];
#pragma unroll
        for (int ct = 0; ct < 4; ++ct) kbt[ct] = load_B_nk(kh + (ct * 16) * HD, HD, lane);
#pragma unroll
        for (int kk2 = 0; kk2 < 2; ++kk2)
#pragma unroll
            for (int dt = 0; dt < 2; ++dt)
                vbt[kk2][dt] = load_B_kn(vh + (kk2 * 32) * HD + dt * 16, HD, lane);

        for (int mt = 0; mt < 4; ++mt) {
            // S row-block 16x64 : 4 tiles, K = head_dim = 32 (single wmma each)
            v16bf qa = load_A_lds(qh + (mt * 16) * HD, HD, lane);
            v8f sacc[4];
#pragma unroll
            for (int ct = 0; ct < 4; ++ct) sacc[ct] = wmma_bf16(qa, kbt[ct], vzero8());

            // + relative-position bias (gather) + shifted-window mask (LDS)
#pragma unroll
            for (int r = 0; r < 8; ++r) {
                const int nq = mt * 16 + r + 8 * half;
                const int y1 = nq >> 3, x1 = nq & 7;
#pragma unroll
                for (int ct = 0; ct < 4; ++ct) {
                    const int nk = ct * 16 + nloc;
                    const int idx = (y1 - (nk >> 3) + 7) * 15 + (x1 - (nk & 7) + 7);
                    sacc[ct][r] += rel_table[idx * HEADS + h] + mk[nq * 64 + nk];
                }
            }

            // row softmax: reduce across 4 reg-tiles + 16 lanes (xor 1/2/4/8 stays in half)
#pragma unroll
            for (int r = 0; r < 8; ++r) {
                float mx = fmaxf(fmaxf(sacc[0][r], sacc[1][r]), fmaxf(sacc[2][r], sacc[3][r]));
                mx = fmaxf(mx, __shfl_xor(mx, 1));
                mx = fmaxf(mx, __shfl_xor(mx, 2));
                mx = fmaxf(mx, __shfl_xor(mx, 4));
                mx = fmaxf(mx, __shfl_xor(mx, 8));
                float sum = 0.0f;
#pragma unroll
                for (int ct = 0; ct < 4; ++ct) {
                    const float e = __expf(sacc[ct][r] - mx);
                    sacc[ct][r] = e;
                    sum += e;
                }
                sum += __shfl_xor(sum, 1);
                sum += __shfl_xor(sum, 2);
                sum += __shfl_xor(sum, 4);
                sum += __shfl_xor(sum, 8);
                const float inv = 1.0f / sum;
#pragma unroll
                for (int ct = 0; ct < 4; ++ct)
                    ph[(r + 8 * half) * 64 + ct * 16 + nloc] = (__bf16)(sacc[ct][r] * inv);
            }
            // wave-private LDS round trip (C-layout -> A-layout); DS is in-order per wave
            asm volatile("s_wait_dscnt 0" ::: "memory");

            // O = P V : 16x32, K = 64 keys (two k-steps)
            v8f o0 = vzero8(), o1 = vzero8();
#pragma unroll
            for (int kk2 = 0; kk2 < 2; ++kk2) {
                v16bf pa = load_A_lds(ph + kk2 * 32, 64, lane);
                o0 = wmma_bf16(pa, vbt[kk2][0], o0);
                o1 = wmma_bf16(pa, vbt[kk2][1], o1);
            }
            // y[m][h*32 + d]  (xs region reused)
#pragma unroll
            for (int r = 0; r < 8; ++r) {
                const int m = mt * 16 + r + 8 * half;
                xs[m * DIMC + h * HD + nloc]      = (__bf16)o0[r];
                xs[m * DIMC + h * HD + 16 + nloc] = (__bf16)o1[r];
            }
        }
    }
    __syncthreads();

    // ---- proj GEMM: [64 x 256] @ [256 x 256] ; wave wv owns output cols [wv*32, wv*32+32) ----
    float* ow = out + (size_t)w * (N_TOK * DIMC);
    for (int ctile = 0; ctile < 2; ++ctile) {
        const int ob = wv * 32 + ctile * 16;
        v8f acc[4];
#pragma unroll
        for (int mt = 0; mt < 4; ++mt) acc[mt] = vzero8();
        for (int ks = 0; ks < 8; ++ks) {
            const int kb = ks * 32;
            v16bf b = load_B_nk_f32(proj_w + (size_t)ob * DIMC + kb, DIMC, lane);
#pragma unroll
            for (int mt = 0; mt < 4; ++mt) {
                v16bf a = load_A_lds(xs + (mt * 16) * DIMC + kb, DIMC, lane);
                acc[mt] = wmma_bf16(a, b, acc[mt]);
            }
        }
        const int o = ob + nloc;
        const float pb = proj_b[o];
#pragma unroll
        for (int mt = 0; mt < 4; ++mt)
#pragma unroll
            for (int r = 0; r < 8; ++r) {
                const int m = mt * 16 + r + 8 * half;
                ow[(size_t)m * DIMC + o] = acc[mt][r] + pb;
            }
    }
}

extern "C" void kernel_launch(void* const* d_in, const int* in_sizes, int n_in,
                              void* d_out, int out_size, void* d_ws, size_t ws_size,
                              hipStream_t stream) {
    const float* x         = (const float*)d_in[0];
    const float* attn_mask = (const float*)d_in[1];
    const float* qkv_w     = (const float*)d_in[2];
    const float* qkv_b     = (const float*)d_in[3];
    const float* rel_table = (const float*)d_in[4];
    const float* proj_w    = (const float*)d_in[5];
    const float* proj_b    = (const float*)d_in[6];
    float* out             = (float*)d_out;

    const int W_B = in_sizes[0] / (N_TOK * DIMC);          // 8192 windows
    const int nW  = in_sizes[1] / (N_TOK * N_TOK);         // 4096 masks

    const size_t shmem = 163840;  // 160 KB carved LDS (within 320 KB/WGP)
    (void)hipFuncSetAttribute((const void*)winattn_fused,
                              hipFuncAttributeMaxDynamicSharedMemorySize, (int)shmem);

    winattn_fused<<<dim3(W_B), dim3(256), shmem, stream>>>(
        x, attn_mask, qkv_w, qkv_b, rel_table, proj_w, proj_b, out, nW);
}
